// DS_Attention_7636451852327
// MI455X (gfx1250) — matrix operand order repaired and checked by reading, not compiled, not run on hardware
//
#include <hip/hip_runtime.h>

// ---------------------------------------------------------------------------
// DS_Attention on gfx1250 (MI455X): three bf16-WMMA GEMMs + tiny per-frame
// attention. fp32 inputs converted to bf16 at staging; f32 accumulate.
// Double-buffered LDS, padded strides for conflict-free fragment loads.
// ---------------------------------------------------------------------------

typedef __bf16 bf16_t;
typedef __attribute__((ext_vector_type(16))) __bf16 v16bf;
typedef __attribute__((ext_vector_type(8)))  __bf16 v8bf;
typedef __attribute__((ext_vector_type(4)))  __bf16 v4bf;
typedef __attribute__((ext_vector_type(8)))  float  v8f;

#define BM  128
#define BN  128
#define BK  32
#define BKP 40   // padded K stride (elements): bank-conflict-free ds_load_b128

// ---- WMMA fragment loaders (layouts per CDNA5 ISA §7.12.2) -----

// A tile in LDS: [BM][BKP] bf16. 16-bit A 16x32 layout:
// lanes 0-15: M=lane, K in {0..7,16..23}; lanes 16-31: M=lane-16, K in {8..15,24..31}.
__device__ __forceinline__ v16bf load_a_frag(const bf16_t* As, int rowBase, int lane) {
    const int m    = lane & 15;
    const int half = (lane >> 4) & 1;
    const bf16_t* p = As + (rowBase + m) * BKP + half * 8;
    v8bf lo = *(const v8bf*)(p);        // K = half*8 + 0..7
    v8bf hi = *(const v8bf*)(p + 16);   // K = 16 + half*8 + 0..7
    v16bf r;
#pragma unroll
    for (int i = 0; i < 8; ++i) { r[i] = lo[i]; r[8 + i] = hi[i]; }
    return r;
}

// B tile in LDS pre-transposed: [BN][BKP] bf16 ([n][k]). 16-bit B 32x16:
// lanes 0-15: N=lane, K in {0..7,16..23}; lanes 16-31: N=lane-16, K in {8..15,24..31}.
__device__ __forceinline__ v16bf load_b_frag(const bf16_t* Bs, int colBase, int lane) {
    const int n  = lane & 15;
    const int kh = (lane >> 4) & 1;
    const bf16_t* p = Bs + (colBase + n) * BKP + kh * 8;
    v8bf lo = *(const v8bf*)(p);
    v8bf hi = *(const v8bf*)(p + 16);
    v16bf r;
#pragma unroll
    for (int i = 0; i < 8; ++i) { r[i] = lo[i]; r[8 + i] = hi[i]; }
    return r;
}

// ---- Generic bf16 WMMA GEMM: C[M,N] = A[M,K(row lda)] * B[K,N(row ldb)] ---
// Kb = valid rows of B (rows in [Kb,K) read as zero). M,K multiples of BM/BK.
__global__ __launch_bounds__(256) void wmma_gemm_bf16(
    const float* __restrict__ A, const float* __restrict__ B,
    float* __restrict__ C, const float* __restrict__ bias,
    int M, int N, int K, int Kb, int lda, int ldb, int ldc)
{
    __shared__ __align__(16) bf16_t As[2][BM][BKP];
    __shared__ __align__(16) bf16_t Bs[2][BN][BKP];

    const int tid   = threadIdx.x;
    const int lane  = tid & 31;
    const int wave  = tid >> 5;          // 0..7
    const int waveM = wave >> 1;         // 0..3 -> 32-row slab
    const int waveN = wave & 1;          // 0..1 -> 64-col slab
    const int rbase = waveM * 32;
    const int cbase = waveN * 64;

    const int blockRow = blockIdx.y * BM;
    const int blockCol = blockIdx.x * BN;
    const bool fullN   = (blockCol + BN <= N);   // uniform per block

    // staging: A -> 2 threads/row x 16 floats; B -> 4 k-rows x 4 cols/thread
    const int arow = tid >> 1;           // 0..127
    const int ak0  = (tid & 1) * 16;     // 0 / 16
    const int bn0  = (tid & 31) * 4;     // 0..124 (column group)
    const int bk0  = (tid >> 5) * 4;     // 0..28  (k group)
    const int bcol = blockCol + bn0;

    v8f acc[2][4];
#pragma unroll
    for (int i = 0; i < 2; ++i)
#pragma unroll
        for (int j = 0; j < 4; ++j)
            acc[i][j] = (v8f){0.f, 0.f, 0.f, 0.f, 0.f, 0.f, 0.f, 0.f};

    float4 ar[4];     // A staging regs: 16 contiguous k
    float4 br[4];     // B staging regs: 4 k-rows x 4 cols

    auto load_regs = [&](int k0) {
        const float* ga = A + (size_t)(blockRow + arow) * lda + (k0 + ak0);
#pragma unroll
        for (int i = 0; i < 4; ++i) ar[i] = ((const float4*)ga)[i];
        if (fullN && (k0 + BK <= Kb)) {
#pragma unroll
            for (int j = 0; j < 4; ++j)
                br[j] = *(const float4*)(B + (size_t)(k0 + bk0 + j) * ldb + bcol);
        } else {
#pragma unroll
            for (int j = 0; j < 4; ++j) {
                const int kk = k0 + bk0 + j;
                const bool kok = (kk < Kb);
                float* bf = (float*)&br[j];
#pragma unroll
                for (int i = 0; i < 4; ++i) {
                    const int col = bcol + i;
                    bf[i] = (kok && col < N) ? B[(size_t)kk * ldb + col] : 0.0f;
                }
            }
        }
    };

    auto store_stage = [&](int buf) {
        const float* af = (const float*)ar;
        v8bf p0, p1;
#pragma unroll
        for (int i = 0; i < 8; ++i) { p0[i] = (bf16_t)af[i]; p1[i] = (bf16_t)af[8 + i]; }
        *(v8bf*)&As[buf][arow][ak0]     = p0;
        *(v8bf*)&As[buf][arow][ak0 + 8] = p1;
        const float* bf = (const float*)br;
#pragma unroll
        for (int i = 0; i < 4; ++i) {    // pack 4 k-values per column -> b64
            v4bf p = { (bf16_t)bf[0 * 4 + i], (bf16_t)bf[1 * 4 + i],
                       (bf16_t)bf[2 * 4 + i], (bf16_t)bf[3 * 4 + i] };
            *(v4bf*)&Bs[buf][bn0 + i][bk0] = p;
        }
    };

    const int nk = K / BK;
    load_regs(0);
    store_stage(0);

    for (int kt = 0; kt < nk; ++kt) {
        const int cur = kt & 1;
        __syncthreads();
        if (kt + 1 < nk) load_regs((kt + 1) * BK);   // overlap with WMMA below

        v16bf af0 = load_a_frag(&As[cur][0][0], rbase,      lane);
        v16bf af1 = load_a_frag(&As[cur][0][0], rbase + 16, lane);
#pragma unroll
        for (int cn = 0; cn < 4; ++cn) {
            v16bf bfr = load_b_frag(&Bs[cur][0][0], cbase + cn * 16, lane);
            acc[0][cn] = __builtin_amdgcn_wmma_f32_16x16x32_bf16(
                false, af0, false, bfr, (short)0, acc[0][cn], false, false);
            acc[1][cn] = __builtin_amdgcn_wmma_f32_16x16x32_bf16(
                false, af1, false, bfr, (short)0, acc[1][cn], false, false);
        }

        if (kt + 1 < nk) store_stage(cur ^ 1);
    }

    // --- store C: f32 16x16 C/D layout: elem r <-> M = r + 8*lane[4], N = lane[3:0]
    const int nl = lane & 15;
    const int mh = (lane >> 4) * 8;
#pragma unroll
    for (int rm = 0; rm < 2; ++rm) {
#pragma unroll
        for (int cn = 0; cn < 4; ++cn) {
            const int col = blockCol + cbase + cn * 16 + nl;
            if (col < N) {
                const float badd = bias ? bias[col] : 0.0f;
                const int row = blockRow + rbase + rm * 16 + mh;
#pragma unroll
                for (int r = 0; r < 8; ++r)
                    C[(size_t)(row + r) * ldc + col] = acc[rm][cn][r] + badd;
            }
        }
    }
}

// ---------------------------------------------------------------------------
// Tiny per-frame attention: one block per (b,n), one wave per head.
// qk: [16384][2352] (q | k, each [8 heads][147]); v: [16384][1184] in/out.
// ---------------------------------------------------------------------------
__global__ __launch_bounds__(256) void frame_attn(
    const float* __restrict__ qk, float* __restrict__ v)
{
    const int bn   = blockIdx.x;
    const int tid  = threadIdx.x;
    const int lane = tid & 31;
    const int h    = tid >> 5;          // head 0..7

    __shared__ float ka[8][24][6];
    __shared__ float va[8][24][6];
    __shared__ float pr[8][24][24];

    const size_t qrow = (size_t)bn * 2352;
    const size_t vrow = (size_t)bn * 1184;
    const int    hb   = h * 147;

    for (int i = lane; i < 144; i += 32) {
        (&ka[h][0][0])[i] = qk[qrow + 1176 + hb + 3 + i];
        (&va[h][0][0])[i] = v[vrow + hb + 3 + i];
    }
    float qr[6];
    if (lane < 24) {
#pragma unroll
        for (int e = 0; e < 6; ++e)
            qr[e] = qk[qrow + hb + 3 + lane * 6 + e];
    }
    __syncthreads();

    if (lane < 24) {
        float s[24];
        float mx = -3.0e38f;
#pragma unroll
        for (int j = 0; j < 24; ++j) {
            float d = 0.f;
#pragma unroll
            for (int e = 0; e < 6; ++e) d += qr[e] * ka[h][j][e];
            d *= 0.4082482904638631f;   // 1/sqrt(6)
            s[j] = d;
            mx = fmaxf(mx, d);
        }
        float sum = 0.f;
#pragma unroll
        for (int j = 0; j < 24; ++j) { s[j] = __expf(s[j] - mx); sum += s[j]; }
        const float inv = 1.0f / sum;
#pragma unroll
        for (int j = 0; j < 24; ++j) pr[h][lane][j] = s[j] * inv;
    }
    __syncthreads();

    if (lane == 0) {   // sequential custom weighting chain (row 0 / col 0)
        float (*P)[24] = pr[h];
        P[0][6]  = (P[0][6]  + P[0][3])  * 0.5f;  P[6][0]  = (P[6][0]  + P[3][0])  * 0.5f;
        P[0][9]  = (P[0][9]  + P[0][6])  * 0.5f;  P[9][0]  = (P[9][0]  + P[6][0])  * 0.5f;
        P[0][12] = (P[0][12] + P[0][9])  * 0.5f;  P[12][0] = (P[12][0] + P[9][0])  * 0.5f;
        P[0][13] = (P[0][13] + P[0][9])  * 0.5f;  P[13][0] = (P[13][0] + P[9][0])  * 0.5f;
        P[0][14] = (P[0][14] + P[0][9])  * 0.5f;  P[14][0] = (P[14][0] + P[9][0])  * 0.5f;
        P[0][16] = (P[0][16] + P[0][13]) * 0.5f;  P[16][0] = (P[16][0] + P[13][0]) * 0.5f;
        P[0][17] = (P[0][17] + P[0][14]) * 0.5f;  P[17][0] = (P[17][0] + P[14][0]) * 0.5f;
        P[0][15] = (P[0][15] + P[0][12]) * 0.5f;  P[15][0] = (P[15][0] + P[12][0]) * 0.5f;
    }
    __syncthreads();

    if (lane < 24) {
        float o[6] = {0.f, 0.f, 0.f, 0.f, 0.f, 0.f};
#pragma unroll
        for (int j = 0; j < 24; ++j) {
            const float p = pr[h][lane][j];
#pragma unroll
            for (int e = 0; e < 6; ++e) o[e] += p * va[h][j][e];
        }
#pragma unroll
        for (int e = 0; e < 6; ++e)
            v[vrow + hb + 3 + lane * 6 + e] = o[e];
        // channels 0..2 of each head stay as-is (in-place buffer).
    }
    if (h == 0 && lane < 8)             // zero K-pad columns for GEMM3
        v[vrow + 1176 + lane] = 0.0f;
}

// ---------------------------------------------------------------------------
// Launch. Inputs: query(0) key(1,unused) value(2) qk_w(3) v_w(4) lin_w(5) lin_b(6)
// Workspace: ws_qk [16384][2352] f32, ws_v [16384][1184] f32 (~232 MB total).
// ---------------------------------------------------------------------------
extern "C" void kernel_launch(void* const* d_in, const int* in_sizes, int n_in,
                              void* d_out, int out_size, void* d_ws, size_t ws_size,
                              hipStream_t stream) {
    (void)in_sizes; (void)n_in; (void)out_size; (void)ws_size;
    const float* query = (const float*)d_in[0];
    const float* value = (const float*)d_in[2];
    const float* qk_w  = (const float*)d_in[3];
    const float* v_w   = (const float*)d_in[4];
    const float* lin_w = (const float*)d_in[5];
    const float* lin_b = (const float*)d_in[6];
    float* out = (float*)d_out;

    const int ROWS = 32 * 512;                 // B*N = 16384
    float* ws_qk = (float*)d_ws;
    float* ws_v  = ws_qk + (size_t)ROWS * 2352;

    dim3 blk(256);
    // GEMM1: [16384,512] x [512,2352] -> ws_qk
    wmma_gemm_bf16<<<dim3((2352 + BN - 1) / BN, ROWS / BM), blk, 0, stream>>>(
        query, qk_w, ws_qk, nullptr, ROWS, 2352, 512, 512, 512, 2352, 2352);
    // GEMM2: [16384,512] x [512,1176] -> ws_v (ldc padded to 1184)
    wmma_gemm_bf16<<<dim3((1176 + BN - 1) / BN, ROWS / BM), blk, 0, stream>>>(
        value, v_w, ws_v, nullptr, ROWS, 1176, 512, 512, 512, 1176, 1184);
    // per-frame attention, in place on ws_v
    frame_attn<<<dim3(ROWS), blk, 0, stream>>>(ws_qk, ws_v);
    // GEMM3: [16384,1184(pad)] x [1176,512] + bias -> out
    wmma_gemm_bf16<<<dim3(512 / BN, ROWS / BM), blk, 0, stream>>>(
        ws_v, lin_w, out, lin_b, ROWS, 512, 1184, 1176, 1184, 512, 512);
}
